// SelfAttention_3152505995895
// MI455X (gfx1250) — matrix operand order repaired and checked by reading, not compiled, not run on hardware
//
#include <hip/hip_runtime.h>

typedef _Float16 half_t;
typedef __attribute__((ext_vector_type(4)))  _Float16 v4h;
typedef __attribute__((ext_vector_type(8)))  _Float16 v8h;
typedef __attribute__((ext_vector_type(16))) _Float16 v16h;
typedef __attribute__((ext_vector_type(8)))  float    v8f;
typedef __attribute__((ext_vector_type(4)))  int      v4i;

#define AS1 __attribute__((address_space(1)))
#define AS3 __attribute__((address_space(3)))

static constexpr int N_ = 8, L_ = 1024, H_ = 16, E_ = 1024, D_ = 64;
static constexpr int LDT = 72;                                   // LDS row stride (halves)
static constexpr float SM_C = 1.4426950408889634f / 32.0f;       // log2(e)/sqrt(E)

#if __has_builtin(__builtin_amdgcn_global_load_async_to_lds_b128) && __has_builtin(__builtin_amdgcn_s_wait_asynccnt)
#define ASYNC_BUILTIN 1
#else
#define ASYNC_BUILTIN 0
#endif

static __device__ __forceinline__ v8f wmma_f16(v16h a, v16h b, v8f c) {
  return __builtin_amdgcn_wmma_f32_16x16x32_f16(false, a, false, b, (short)0, c, false, false);
}

// A fragment (16x32 f16): row m = lane&15; K chunks [kb,kb+8) and [kb+16,kb+24), kb = 8*(lane>>4)
static __device__ __forceinline__ v16h frag_a(const half_t* base, int ld, int lane, int koff) {
  int m  = lane & 15;
  int kb = koff + ((lane >> 4) << 3);
  const v8h* p = (const v8h*)(base + m * ld + kb);
  return __builtin_shufflevector(p[0], p[2], 0,1,2,3,4,5,6,7,8,9,10,11,12,13,14,15);
}

// B fragment (32x16 f16) from transposed source: B[k][n] = src[n*ld+k].
// col n = lane&15; K chunks [kb,kb+8) and [kb+16,kb+24), kb = 8*(lane>>4)
static __device__ __forceinline__ v16h frag_bt(const half_t* base, int ld, int lane, int koff) {
  int n  = lane & 15;
  int kb = koff + ((lane >> 4) << 3);
  const v8h* p = (const v8h*)(base + n * ld + kb);
  return __builtin_shufflevector(p[0], p[2], 0,1,2,3,4,5,6,7,8,9,10,11,12,13,14,15);
}

static __device__ __forceinline__ float redmax16(float v) {
  v = fmaxf(v, __shfl_xor(v, 1));
  v = fmaxf(v, __shfl_xor(v, 2));
  v = fmaxf(v, __shfl_xor(v, 4));
  v = fmaxf(v, __shfl_xor(v, 8));
  return v;
}
static __device__ __forceinline__ float redsum16(float v) {
  v += __shfl_xor(v, 1); v += __shfl_xor(v, 2);
  v += __shfl_xor(v, 4); v += __shfl_xor(v, 8);
  return v;
}

// ---- async tile staging: 64 rows x 64 halves, global row stride grs -> LDS stride LDT
static __device__ __forceinline__ void stage64x64_async(half_t* dst, const half_t* src,
                                                        long grs, int tid) {
  int r = tid >> 1, c = (tid & 1) * 32;
  const half_t* g = src + (long)r * grs + c;
  half_t* l = dst + r * LDT + c;
#if ASYNC_BUILTIN
#pragma unroll
  for (int u = 0; u < 4; ++u)
    __builtin_amdgcn_global_load_async_to_lds_b128((AS1 v4i*)(g + u * 8),
                                                   (AS3 v4i*)(l + u * 8), 0, 0);
#else
#pragma unroll
  for (int u = 0; u < 4; ++u) {
    unsigned lo = (unsigned)(uintptr_t)(AS3 half_t*)(l + u * 8);
    asm volatile("global_load_async_to_lds_b128 %0, %1, off"
                 :: "v"(lo), "v"(g + u * 8) : "memory");
  }
#endif
}
static __device__ __forceinline__ void stage_wait() {
#if ASYNC_BUILTIN
  __builtin_amdgcn_s_wait_asynccnt(0);
#else
  asm volatile("s_wait_asynccnt 0x0" ::: "memory");
#endif
}

// ---- transposed tile staging (sync): dstT[d][k] = src[k][d], 64x64 halves
static __device__ __forceinline__ void stage64x64_T(half_t* dstT, const half_t* src,
                                                    long grs, int tid) {
  int r = tid >> 1, c = (tid & 1) * 32;
  const v8h* gs = (const v8h*)(src + (long)r * grs + c);
#pragma unroll
  for (int u = 0; u < 4; ++u) {
    v8h d = gs[u];
#pragma unroll
    for (int e = 0; e < 8; ++e)
      dstT[(c + u * 8 + e) * LDT + r] = d[e];
  }
}

// ---------------------------------------------------------------- QKV projection
__global__ __launch_bounds__(128) void k_qkv(
    const float* __restrict__ x,
    const float* __restrict__ Wq, const float* __restrict__ bq,
    const float* __restrict__ Wk, const float* __restrict__ bk,
    const float* __restrict__ Wv, const float* __restrict__ bv,
    half_t* __restrict__ qh, half_t* __restrict__ kh, half_t* __restrict__ vh) {
  __shared__ half_t zt[64 * LDT];
  __shared__ half_t wt[3 * 64 * LDT];
  const int tid = threadIdx.x, lane = tid & 31, wid = tid >> 5;
  const int nlane = lane & 15;
  const long row0 = (long)blockIdx.x * 64;
  const float* Ws[3] = {Wq, Wk, Wv};
#pragma unroll
  for (int m = 0; m < 3; ++m) {
#pragma unroll
    for (int j = 0; j < 8; ++j) {
      int idx = tid + j * 128;
      int r = idx >> 4, c4 = (idx & 15) << 2;
      float4 f = *(const float4*)(Ws[m] + r * 64 + c4);
      v4h h4 = {(half_t)f.x, (half_t)f.y, (half_t)f.z, (half_t)f.w};
      *(v4h*)&wt[m * 64 * LDT + r * LDT + c4] = h4;
    }
  }
#pragma unroll
  for (int j = 0; j < 8; ++j) {
    int idx = tid + j * 128;
    int r = idx >> 4, c4 = (idx & 15) << 2;
    float4 f = *(const float4*)(x + (row0 + r) * 64 + c4);
    v4h h4 = {(half_t)f.x, (half_t)f.y, (half_t)f.z, (half_t)f.w};
    *(v4h*)&zt[r * LDT + c4] = h4;
  }
  __syncthreads();
  v16h a0 = frag_a(&zt[wid * 16 * LDT], LDT, lane, 0);
  v16h a1 = frag_a(&zt[wid * 16 * LDT], LDT, lane, 32);
  const float* bs[3]   = {bq, bk, bv};
  half_t*      outs[3] = {qh, kh, vh};
#pragma unroll
  for (int m = 0; m < 3; ++m) {
#pragma unroll
    for (int ns = 0; ns < 4; ++ns) {
      v8f c = {};
      c = wmma_f16(a0, frag_bt(&wt[m * 64 * LDT + ns * 16 * LDT], LDT, lane, 0),  c);
      c = wmma_f16(a1, frag_bt(&wt[m * 64 * LDT + ns * 16 * LDT], LDT, lane, 32), c);
      float bb = bs[m][ns * 16 + nlane];
      long mg = row0 + wid * 16 + ((lane >> 4) << 3);
#pragma unroll
      for (int r = 0; r < 8; ++r)
        outs[m][(mg + r) * 64 + ns * 16 + nlane] = (half_t)(c[r] + bb);
    }
  }
}

// ---------------------------------------------------------------- Erel f32 -> f16
__global__ void k_cvt_erel(const float* __restrict__ e, half_t* __restrict__ eh) {
  int i = blockIdx.x * 256 + threadIdx.x;
  eh[i] = (half_t)e[i];
}

// ---------------------------------------------------------------- fused flash attention (softmax part)
__global__ __launch_bounds__(128) void k_attn(
    const half_t* __restrict__ qh, const half_t* __restrict__ kh,
    const half_t* __restrict__ vh, half_t* __restrict__ ah) {
  __shared__ half_t ktile[2][64 * LDT];
  __shared__ half_t vtT[2][64 * LDT];
  __shared__ half_t pbuf[4 * 16 * LDT];
  const int tid = threadIdx.x, lane = tid & 31, wid = tid >> 5;
  const int q0 = blockIdx.x * 64, hh = blockIdx.y, nn = blockIdx.z;
  const int nlane = lane & 15, hl = lane >> 4;

  const half_t* qbase = qh + (((long)(nn * L_ + q0 + wid * 16)) * H_ + hh) * D_;
  v16h aq0 = frag_a(qbase, H_ * D_, lane, 0);
  v16h aq1 = frag_a(qbase, H_ * D_, lane, 32);

  v8f o[4] = {};
  float mi[8], li[8];
#pragma unroll
  for (int r = 0; r < 8; ++r) { mi[r] = -3.0e38f; li[r] = 0.f; }
  half_t* pw = &pbuf[wid * 16 * LDT];

  const long krs = (long)H_ * D_;
  const long kvbase = ((long)nn * L_) * krs + hh * D_;
  stage64x64_async(ktile[0], kh + kvbase, krs, tid);
  stage64x64_T(vtT[0], vh + kvbase, krs, tid);

  for (int kti = 0; kti < L_ / 64; ++kti) {
    stage_wait();
    __syncthreads();                       // double-buffered: single barrier per tile
    const int cur = kti & 1;
    if (kti + 1 < L_ / 64) {
      const long nb = kvbase + (long)(kti + 1) * 64 * krs;
      stage64x64_async(ktile[cur ^ 1], kh + nb, krs, tid);
      stage64x64_T(vtT[cur ^ 1], vh + nb, krs, tid);
    }
    const half_t* kt = ktile[cur];
    const half_t* vt = vtT[cur];
    v8f s[4];
#pragma unroll
    for (int j = 0; j < 4; ++j) {
      v8f c = {};
      c = wmma_f16(aq0, frag_bt(&kt[j * 16 * LDT], LDT, lane, 0),  c);
      c = wmma_f16(aq1, frag_bt(&kt[j * 16 * LDT], LDT, lane, 32), c);
      s[j] = c;
    }
    // online softmax per row (rows hl*8+r live in this lane's 16-lane half)
#pragma unroll
    for (int r = 0; r < 8; ++r) {
      float mx = fmaxf(fmaxf(s[0][r], s[1][r]), fmaxf(s[2][r], s[3][r]));
      mx = redmax16(mx);
      float mn = fmaxf(mi[r], mx);
      float scale = __builtin_amdgcn_exp2f((mi[r] - mn) * SM_C);
      mi[r] = mn;
      float rs = 0.f;
#pragma unroll
      for (int j = 0; j < 4; ++j) {
        float p = __builtin_amdgcn_exp2f((s[j][r] - mn) * SM_C);
        s[j][r] = p;
        rs += p;
      }
      rs = redsum16(rs);
      li[r] = li[r] * scale + rs;
#pragma unroll
      for (int j = 0; j < 4; ++j) o[j][r] *= scale;
    }
    // P (C-layout) -> LDS -> A fragments
#pragma unroll
    for (int j = 0; j < 4; ++j)
#pragma unroll
      for (int r = 0; r < 8; ++r)
        pw[(hl * 8 + r) * LDT + j * 16 + nlane] = (half_t)s[j][r];
    v16h ap0 = frag_a(pw, LDT, lane, 0);
    v16h ap1 = frag_a(pw, LDT, lane, 32);
#pragma unroll
    for (int j = 0; j < 4; ++j) {
      o[j] = wmma_f16(ap0, frag_bt(&vt[j * 16 * LDT], LDT, lane, 0),  o[j]);
      o[j] = wmma_f16(ap1, frag_bt(&vt[j * 16 * LDT], LDT, lane, 32), o[j]);
    }
  }
#pragma unroll
  for (int r = 0; r < 8; ++r) {
    float inv = 1.0f / li[r];
    long qrow = (long)nn * L_ + q0 + wid * 16 + hl * 8 + r;
#pragma unroll
    for (int j = 0; j < 4; ++j)
      ah[qrow * E_ + hh * D_ + j * 16 + nlane] = (half_t)(o[j][r] * inv);
  }
}

// ---------------------------------------------------------------- relative (skew) term: S.V added post-softmax
__global__ __launch_bounds__(128) void k_rel(
    const half_t* __restrict__ qh, const half_t* __restrict__ vh,
    const half_t* __restrict__ eh, half_t* __restrict__ ah) {
  __shared__ half_t vtT[64 * LDT];
  __shared__ half_t eb[130 * LDT];       // Erel band (2 pad rows for the unread G col 31)
  __shared__ half_t pbuf[4 * 16 * LDT];
  __shared__ float  gb[4 * 16 * 32];
  const int tid = threadIdx.x, lane = tid & 31, wid = tid >> 5;
  const int q0 = blockIdx.x * 64, hh = blockIdx.y, nn = blockIdx.z;
  const int nlane = lane & 15, hl = lane >> 4;
  const int qs = q0 + wid * 16;

  const half_t* qbase = qh + (((long)(nn * L_ + qs)) * H_ + hh) * D_;
  v16h aq0 = frag_a(qbase, H_ * D_, lane, 0);
  v16h aq1 = frag_a(qbase, H_ * D_, lane, 32);
  v8f o[4] = {};
  half_t* pw = &pbuf[wid * 16 * LDT];
  float*  gw = &gb[wid * 16 * 32];
  const long krs = (long)H_ * D_;
  const int nkt = q0 / 64 + 1;

  for (int kti = 0; kti < nkt; ++kti) {
    const int k0 = kti * 64;
    const int c00 = (L_ - 1) + k0 - q0;
    stage64x64_T(vtT, vh + ((long)(nn * L_ + k0)) * krs + hh * D_, krs, tid);
    {
      int r = tid;                       // one Erel row per thread
      int t = c00 - 64 + r;
      v8h* ed = (v8h*)&eb[r * LDT];
      if (t >= 0 && t < L_) {
        const v8h* es = (const v8h*)(eh + (long)t * D_);
#pragma unroll
        for (int u = 0; u < 8; ++u) ed[u] = es[u];
      } else {
        v8h z = {};
#pragma unroll
        for (int u = 0; u < 8; ++u) ed[u] = z;
      }
    }
    __syncthreads();
#pragma unroll
    for (int j = 0; j < 4; ++j) {
      const int ks = k0 + j * 16;
      if (ks <= qs + 15) {               // wave-uniform; EXEC stays all-ones for WMMA
        const int rb = 49 + 16 * (j - wid);   // eb row of Erel[t_base]
        v8f g0 = {}, g1 = {};
        g0 = wmma_f16(aq0, frag_bt(&eb[rb * LDT], LDT, lane, 0),  g0);
        g0 = wmma_f16(aq1, frag_bt(&eb[rb * LDT], LDT, lane, 32), g0);
        g1 = wmma_f16(aq0, frag_bt(&eb[(rb + 16) * LDT], LDT, lane, 0),  g1);
        g1 = wmma_f16(aq1, frag_bt(&eb[(rb + 16) * LDT], LDT, lane, 32), g1);
#pragma unroll
        for (int r = 0; r < 8; ++r) {
          gw[(hl * 8 + r) * 32 + nlane]      = g0[r];
          gw[(hl * 8 + r) * 32 + 16 + nlane] = g1[r];
        }
        // S[i][jj] = G[i][15+jj-i], masked to k<=q (skew of masked QE)
        const int i = nlane;
#pragma unroll
        for (int e = 0; e < 8; ++e) {
          int jj = hl * 8 + e;
          float v = 0.f;
          if (jj - i <= qs - ks) v = gw[i * 32 + 15 + jj - i];
          pw[i * LDT + j * 16 + jj] = (half_t)v;
        }
      } else {
#pragma unroll
        for (int e = 0; e < 8; ++e)
          pw[nlane * LDT + j * 16 + hl * 8 + e] = (half_t)0.f;
      }
    }
    v16h ap0 = frag_a(pw, LDT, lane, 0);
    v16h ap1 = frag_a(pw, LDT, lane, 32);
#pragma unroll
    for (int j = 0; j < 4; ++j) {
      o[j] = wmma_f16(ap0, frag_bt(&vtT[j * 16 * LDT], LDT, lane, 0),  o[j]);
      o[j] = wmma_f16(ap1, frag_bt(&vtT[j * 16 * LDT], LDT, lane, 32), o[j]);
    }
    __syncthreads();
  }
#pragma unroll
  for (int r = 0; r < 8; ++r) {
    long qrow = (long)nn * L_ + q0 + wid * 16 + hl * 8 + r;
#pragma unroll
    for (int j = 0; j < 4; ++j) {
      long idx = qrow * E_ + hh * D_ + j * 16 + nlane;
      ah[idx] = (half_t)((float)ah[idx] + o[j][r]);
    }
  }
}

// ---------------------------------------------------------------- output projection GEMM (8192x1024x1024)
static __device__ __forceinline__ void stage_wo(half_t* dst, const float* src, int tid) {
#pragma unroll
  for (int j = 0; j < 8; ++j) {
    int idx = tid + j * 128;
    int r = idx >> 4, c4 = (idx & 15) << 2;
    float4 f = *(const float4*)(src + (long)r * E_ + c4);
    v4h h4 = {(half_t)f.x, (half_t)f.y, (half_t)f.z, (half_t)f.w};
    *(v4h*)&dst[r * LDT + c4] = h4;
  }
}

__global__ __launch_bounds__(128) void k_out(
    const half_t* __restrict__ ah, const float* __restrict__ Wo,
    const float* __restrict__ bo, float* __restrict__ out) {
  __shared__ half_t at[2][64 * LDT];
  __shared__ half_t wt[2][64 * LDT];
  const int tid = threadIdx.x, lane = tid & 31, wid = tid >> 5;
  const int n0 = blockIdx.x * 64;
  const long m0 = (long)blockIdx.y * 64;
  const int nlane = lane & 15, hl = lane >> 4;
  v8f c[4] = {};
  stage64x64_async(at[0], ah + m0 * E_, E_, tid);
  stage_wo(wt[0], Wo + (long)n0 * E_, tid);
  for (int ks = 0; ks < E_ / 64; ++ks) {
    stage_wait();
    __syncthreads();
    const int cur = ks & 1;
    if (ks + 1 < E_ / 64) {
      stage64x64_async(at[cur ^ 1], ah + m0 * E_ + (ks + 1) * 64, E_, tid);
      stage_wo(wt[cur ^ 1], Wo + (long)n0 * E_ + (ks + 1) * 64, tid);
    }
    v16h a0 = frag_a(&at[cur][wid * 16 * LDT], LDT, lane, 0);
    v16h a1 = frag_a(&at[cur][wid * 16 * LDT], LDT, lane, 32);
#pragma unroll
    for (int ns = 0; ns < 4; ++ns) {
      c[ns] = wmma_f16(a0, frag_bt(&wt[cur][ns * 16 * LDT], LDT, lane, 0),  c[ns]);
      c[ns] = wmma_f16(a1, frag_bt(&wt[cur][ns * 16 * LDT], LDT, lane, 32), c[ns]);
    }
  }
#pragma unroll
  for (int ns = 0; ns < 4; ++ns) {
    float bb = bo[n0 + ns * 16 + nlane];
#pragma unroll
    for (int r = 0; r < 8; ++r)
      out[(m0 + wid * 16 + hl * 8 + r) * E_ + n0 + ns * 16 + nlane] = c[ns][r] + bb;
  }
}

extern "C" void kernel_launch(void* const* d_in, const int* in_sizes, int n_in,
                              void* d_out, int out_size, void* d_ws, size_t ws_size,
                              hipStream_t stream) {
  (void)in_sizes; (void)n_in; (void)out_size; (void)ws_size;
  const float* x    = (const float*)d_in[0];
  const float* Wq   = (const float*)d_in[1];
  const float* bq   = (const float*)d_in[2];
  const float* Wk   = (const float*)d_in[3];
  const float* bk   = (const float*)d_in[4];
  const float* Wv   = (const float*)d_in[5];
  const float* bv   = (const float*)d_in[6];
  const float* Erel = (const float*)d_in[7];
  const float* Wo   = (const float*)d_in[8];
  const float* bo   = (const float*)d_in[9];
  float* out = (float*)d_out;

  const size_t NLHD = (size_t)N_ * L_ * H_ * D_;       // 8.4M elems
  half_t* qh = (half_t*)d_ws;
  half_t* kh = qh + NLHD;
  half_t* vh = kh + NLHD;
  half_t* eh = vh + NLHD;
  half_t* ah = eh + (size_t)L_ * D_;                   // total ws ~64 MB (f16)

  k_qkv<<<(N_ * L_ * H_) / 64, 128, 0, stream>>>(x, Wq, bq, Wk, bk, Wv, bv, qh, kh, vh);
  k_cvt_erel<<<(L_ * D_) / 256, 256, 0, stream>>>(Erel, eh);
  dim3 g3(L_ / 64, H_, N_);
  k_attn<<<g3, 128, 0, stream>>>(qh, kh, vh, ah);
  k_rel<<<g3, 128, 0, stream>>>(qh, vh, eh, ah);
  dim3 g5(E_ / 64, (N_ * L_) / 64);
  k_out<<<g5, 128, 0, stream>>>(ah, Wo, bo, out);
}
